// Attention_15676630631260
// MI455X (gfx1250) — compile-verified, matrix operand order checked by reading, not run on hardware
//
#include <hip/hip_runtime.h>
#include <hip/hip_bf16.h>

typedef __attribute__((ext_vector_type(16))) _Float16 v16h;
typedef __attribute__((ext_vector_type(8)))  _Float16 v8h;
typedef __attribute__((ext_vector_type(8)))  float    v8f;
typedef __attribute__((ext_vector_type(4)))  float    v4f;

#define B_  2
#define S_  512
#define D_  256
#define P_  128
#define BS_ (B_ * S_)

// ---------------------------------------------------------------------------
// Kernel A: q/k/v = x @ W^T + b   (per-wave 16x16 tile, WMMA f16->f32)
// blockIdx.y selects the matrix (0:q 1:k 2:v) -> block-uniform scalar branch
// in the epilogue instead of per-element exec-mask divergence.
// blockIdx.x covers rowTiles(64) x colTiles(8) = 512 tiles / 8 waves = 64 blocks.
// ---------------------------------------------------------------------------
__global__ __launch_bounds__(256) void qkv_kernel(
    const float* __restrict__ x,
    const float* __restrict__ Wq, const float* __restrict__ bq,
    const float* __restrict__ Wk, const float* __restrict__ bk,
    const float* __restrict__ Wv, const float* __restrict__ bv,
    _Float16* __restrict__ qh, _Float16* __restrict__ kh,
    float* __restrict__ vf)
{
    const int lane = threadIdx.x & 31;
    const int wave = threadIdx.x >> 5;
    const int tile = blockIdx.x * 8 + wave;     // 0..511
    const int mat  = blockIdx.y;                // 0:q 1:k 2:v (block-uniform)

    const int m0  = (tile >> 3) * 16;           // row tile over B*S
    const int p0  = (tile & 7) * 16;            // col tile over P

    const int half = lane >> 4;                 // lane group (K-halves)
    const int mr   = lane & 15;                 // row (A) / col (B) within tile

    const float* W    = (mat == 0) ? Wq : (mat == 1) ? Wk : Wv;
    const float* bias = (mat == 0) ? bq : (mat == 1) ? bk : bv;

    const float* xrow = x + (size_t)(m0 + mr) * D_;   // A: row of x
    const float* wrow = W + (size_t)(p0 + mr) * D_;   // B: row of W == col of W^T

    v8f c = {};
    for (int kb = 0; kb < D_; kb += 32) {
        // A fragment: elements 0..7  -> K = kb + half*8 + e
        //             elements 8..15 -> K = kb + 16 + half*8 + e
        v4f a0 = *(const v4f*)(xrow + kb + half * 8);
        v4f a1 = *(const v4f*)(xrow + kb + half * 8 + 4);
        v4f a2 = *(const v4f*)(xrow + kb + 16 + half * 8);
        v4f a3 = *(const v4f*)(xrow + kb + 16 + half * 8 + 4);
        v4f b0 = *(const v4f*)(wrow + kb + half * 8);
        v4f b1 = *(const v4f*)(wrow + kb + half * 8 + 4);
        v4f b2 = *(const v4f*)(wrow + kb + 16 + half * 8);
        v4f b3 = *(const v4f*)(wrow + kb + 16 + half * 8 + 4);

        v16h a, bf;
#pragma unroll
        for (int e = 0; e < 4; ++e) {
            a[e]      = (_Float16)a0[e];
            a[4 + e]  = (_Float16)a1[e];
            a[8 + e]  = (_Float16)a2[e];
            a[12 + e] = (_Float16)a3[e];
            bf[e]      = (_Float16)b0[e];
            bf[4 + e]  = (_Float16)b1[e];
            bf[8 + e]  = (_Float16)b2[e];
            bf[12 + e] = (_Float16)b3[e];
        }
        c = __builtin_amdgcn_wmma_f32_16x16x32_f16(
                false, a, false, bf, (short)0, c, false, false);
    }

    const float bn = bias[p0 + mr];             // N = mr for this lane
    const int   n  = p0 + mr;

    if (mat == 2) {                             // v -> f32 (scalar branch)
#pragma unroll
        for (int r = 0; r < 8; ++r) {
            const int m = m0 + r + 8 * half;    // C/D: M = vgpr + 8*(lane/16)
            vf[(size_t)m * P_ + n] = c[r] + bn;
        }
    } else {                                    // q/k -> f16
        _Float16* dst = (mat == 0) ? qh : kh;   // scalar select
#pragma unroll
        for (int r = 0; r < 8; ++r) {
            const int m = m0 + r + 8 * half;
            dst[(size_t)m * P_ + n] = (_Float16)(c[r] + bn);
        }
    }
}

// ---------------------------------------------------------------------------
// Kernel B: scores[b,i,j] = sigmoid( (q[b,i,:] . k[b,j,:]) / sqrt(P) )
// WMMA over K=P=128 (4 steps of 32), 128-bit f16 fragment loads.
// tiles: B(2) x it(32) x jt(32) = 2048 tiles, 8 waves per block.
// ---------------------------------------------------------------------------
__global__ __launch_bounds__(256) void score_kernel(
    const _Float16* __restrict__ qh, const _Float16* __restrict__ kh,
    float* __restrict__ scores)
{
    const int lane = threadIdx.x & 31;
    const int wave = threadIdx.x >> 5;
    const int tile = blockIdx.x * 8 + wave;     // 0..2047

    const int b   = tile >> 10;
    const int rem = tile & 1023;
    const int i0  = (rem >> 5) * 16;
    const int j0  = (rem & 31) * 16;

    const int half = lane >> 4;
    const int mr   = lane & 15;

    const _Float16* qrow = qh + (size_t)(b * S_ + i0 + mr) * P_;
    const _Float16* krow = kh + (size_t)(b * S_ + j0 + mr) * P_;

    v8f c = {};
#pragma unroll
    for (int kb = 0; kb < P_; kb += 32) {
        v8h alo = *(const v8h*)(qrow + kb + half * 8);
        v8h ahi = *(const v8h*)(qrow + kb + 16 + half * 8);
        v8h blo = *(const v8h*)(krow + kb + half * 8);
        v8h bhi = *(const v8h*)(krow + kb + 16 + half * 8);
        v16h a, bf;
#pragma unroll
        for (int e = 0; e < 8; ++e) {
            a[e]  = alo[e];  a[8 + e]  = ahi[e];
            bf[e] = blo[e];  bf[8 + e] = bhi[e];
        }
        c = __builtin_amdgcn_wmma_f32_16x16x32_f16(
                false, a, false, bf, (short)0, c, false, false);
    }

    const float inv_sqrt_p = 0.088388347648318447f;   // 1/sqrt(128)
#pragma unroll
    for (int r = 0; r < 8; ++r) {
        const int i = i0 + r + 8 * half;
        const int j = j0 + mr;
        const float z = c[r] * inv_sqrt_p;
        const float s = 1.0f / (1.0f + __expf(-z));
        scores[((size_t)b * S_ + i) * S_ + j] = s;
    }
}

// ---------------------------------------------------------------------------
// Kernel C: out[b,i,j,p] = (j<=i) ? scores[b,i,j] * v[b,j,p] : 0
// Pure streaming store of 256 MB: one float4 per thread, non-temporal.
// scores (2 MB) and v (512 KB) stay L2-resident; this kernel sets the
// runtime floor (~11 us at 23.3 TB/s).
// ---------------------------------------------------------------------------
__global__ __launch_bounds__(256) void out_kernel(
    const float* __restrict__ scores, const float* __restrict__ vf,
    float* __restrict__ out)
{
    const unsigned idx = blockIdx.x * 256u + threadIdx.x; // float4 index
    const unsigned p4  = idx & 31u;                       // 32 float4 = 128 p
    unsigned t = idx >> 5;
    const unsigned j = t & 511u;  t >>= 9;
    const unsigned i = t & 511u;
    const unsigned b = t >> 9;

    const float s = (j <= i) ? scores[((size_t)b * S_ + i) * S_ + j] : 0.0f;
    const v4f v  = ((const v4f*)(vf + ((size_t)b * S_ + j) * P_))[p4];
    v4f r;
    r.x = s * v.x; r.y = s * v.y; r.z = s * v.z; r.w = s * v.w;
    __builtin_nontemporal_store(r, ((v4f*)out) + idx);
}

// ---------------------------------------------------------------------------
extern "C" void kernel_launch(void* const* d_in, const int* in_sizes, int n_in,
                              void* d_out, int out_size, void* d_ws, size_t ws_size,
                              hipStream_t stream)
{
    const float* x  = (const float*)d_in[0];
    const float* Wq = (const float*)d_in[1];
    const float* bq = (const float*)d_in[2];
    const float* Wk = (const float*)d_in[3];
    const float* bk = (const float*)d_in[4];
    const float* Wv = (const float*)d_in[5];
    const float* bv = (const float*)d_in[6];
    float* out = (float*)d_out;

    // workspace layout (16B-aligned segments), 3 MB total
    char* ws = (char*)d_ws;
    _Float16* qh     = (_Float16*)(ws);                    // 256 KB
    _Float16* kh     = (_Float16*)(ws + 256 * 1024);       // 256 KB
    float*    vf     = (float*)   (ws + 512 * 1024);       // 512 KB
    float*    scores = (float*)   (ws + 1024 * 1024);      // 2 MB

    // A: 512 tiles / 8 waves = 64 blocks in x, 3 matrices in y
    qkv_kernel<<<dim3(64, 3), 256, 0, stream>>>(x, Wq, bq, Wk, bk, Wv, bv, qh, kh, vf);
    // B: 2048 tiles / 8 waves = 256 blocks
    score_kernel<<<256, 256, 0, stream>>>(qh, kh, scores);
    // C: 2*512*512*32 float4s = 16,777,216 threads -> 65536 blocks
    out_kernel<<<65536, 256, 0, stream>>>(scores, vf, out);
}
